// Decoder_64080912056533
// MI455X (gfx1250) — matrix operand order repaired and checked by reading, not compiled, not run on hardware
//
#include <hip/hip_runtime.h>
#include <cmath>

// ---------------------------------------------------------------------------
// Problem constants (from reference)
// ---------------------------------------------------------------------------
#define BB   128
#define PP   196
#define EE   2048
#define AA   512
#define DD   512
#define EMBD 512
#define VV   30000
#define LL   25
#define TT   24          // max_dec = L-1

typedef __attribute__((ext_vector_type(16))) __bf16 v16bf;
typedef __attribute__((ext_vector_type(8)))  __bf16 v8bf;
typedef __attribute__((ext_vector_type(8)))  float  v8f;
typedef __bf16 bf16_t;

union AV { v16bf v; v8bf h[2]; };

__device__ __forceinline__ float sigf(float x) { return 1.0f / (1.0f + expf(-x)); }

// ---------------------------------------------------------------------------
// Weight convert + swizzle into the WMMA B-operand lane layout.
// Source: f32 row-major [K, N] (reference stores weights as [in, out]).
// Dest layout: Bp[(((ntile*KCtot + kcBase+kcl)*32 + lane)*16 + e]
//   lane 0-15  -> column ntile*16+lane, K rows kc*32 + 0..7, 16..23
//   lane 16-31 -> column ntile*16+(lane-16), K rows kc*32 + 8..15, 24..31
// so each lane's 16 bf16 (32 B) are contiguous in memory.
// ---------------------------------------------------------------------------
__global__ void swizzleB_kernel(const float* __restrict__ W, bf16_t* __restrict__ Bp,
                                int ldw, int Ntiles, int nkc, int kcBase, int KCtot) {
    long i = (long)blockIdx.x * blockDim.x + threadIdx.x;
    long total = (long)Ntiles * nkc * 32;
    if (i >= total) return;
    int lane = (int)(i & 31);
    int kcl  = (int)((i >> 5) % nkc);
    int nt   = (int)(i / (32L * nkc));
    int koff = (lane < 16) ? 0 : 8;
    int col  = nt * 16 + (lane & 15);
    bf16_t* dst = Bp + (((long)nt * KCtot + kcBase + kcl) * 32 + lane) * 16;
    const float* src = W + (long)(kcl * 32 + koff) * ldw + col;
#pragma unroll
    for (int e = 0; e < 16; ++e)
        dst[e] = (bf16_t)src[(long)(e < 8 ? e : e + 8) * ldw];
}

// f32 -> bf16 (grid-stride)
__global__ void cvt_bf16_kernel(const float* __restrict__ x, bf16_t* __restrict__ y, long n) {
    for (long i = (long)blockIdx.x * blockDim.x + threadIdx.x; i < n;
         i += (long)gridDim.x * blockDim.x)
        y[i] = (bf16_t)x[i];
}

// avg over P; writes bf16 avg (GEMM A operand)
__global__ void mean_kernel(const float* __restrict__ enc, bf16_t* __restrict__ avgbf) {
    long i = (long)blockIdx.x * blockDim.x + threadIdx.x;   // b*E + e
    if (i >= (long)BB * EE) return;
    long b = i >> 11, e = i & (EE - 1);
    const float* p = enc + (b * PP) * (long)EE + e;
    float s = 0.f;
    for (int k = 0; k < PP; ++k) s += p[(long)k * EE];
    avgbf[i] = (bf16_t)(s * (1.0f / PP));
}

// bias_cat = b_ih + b_hh
__global__ void biascat_kernel(const float* a, const float* b, float* o) {
    int i = blockIdx.x * blockDim.x + threadIdx.x;
    if (i < 4 * DD) o[i] = a[i] + b[i];
}

// encoded_captions + dec_len -> d_out tail (cast to float)
__global__ void tail_kernel(const int* caps, const int* lens, float* out_caps, float* out_lens) {
    int i = blockIdx.x * blockDim.x + threadIdx.x;
    if (i < BB * LL) out_caps[i] = (float)caps[i];
    if (i < BB)      out_lens[i] = (float)(lens[i] - 1);
}

// ---------------------------------------------------------------------------
// Generic bf16 WMMA GEMM:  C[M,N] = A[M,K] @ B[K,N] (+bias) (optional row mask)
//   A: bf16 row-major, K % 32 == 0, M % 128 == 0
//   Bp: pre-swizzled bf16 (see above)
//   grid: x = ceil(Ntiles/8), y = M/128 ; block = 256 (8 waves)
//   each wave: one 16-wide N tile, all 8 M tiles of its 128-row block.
//   lens != null => rows are batch indices, zero rows with t >= lens[row]-1.
// ---------------------------------------------------------------------------
__global__ void __launch_bounds__(256)
gemm_wmma_kernel(const bf16_t* __restrict__ A, const bf16_t* __restrict__ Bp,
                 const float* __restrict__ bias, float* __restrict__ C,
                 int K, int KC, int Ntiles, long ldc,
                 const int* __restrict__ lens, int t) {
    int wave  = threadIdx.x >> 5;
    int lane  = threadIdx.x & 31;
    int ntile = blockIdx.x * 8 + wave;
    if (ntile >= Ntiles) return;
    int mbase = blockIdx.y * 128;
    int hi = lane >> 4;          // 0 / 1
    int ml = lane & 15;

    v8f acc[8];
#pragma unroll
    for (int mt = 0; mt < 8; ++mt) acc[mt] = {};

    const bf16_t* bptr = Bp + ((long)ntile * KC * 32 + lane) * 16;
    for (int kc = 0; kc < KC; ++kc) {
        v16bf bv = *(const v16bf*)bptr;           // 32 contiguous bytes
        bptr += 32 * 16;
        const bf16_t* abase = A + (long)(mbase + ml) * K + kc * 32 + hi * 8;
#pragma unroll
        for (int mt = 0; mt < 8; ++mt) {
            const bf16_t* ap = abase + (long)mt * 16 * K;
            AV av;
            av.h[0] = *(const v8bf*)ap;           // K 0-7  (or 8-15)
            av.h[1] = *(const v8bf*)(ap + 16);    // K 16-23 (or 24-31)
            acc[mt] = __builtin_amdgcn_wmma_f32_16x16x32_bf16(
                false, av.v, false, bv, (short)0, acc[mt], false, false);
        }
    }

    int col = ntile * 16 + ml;
    float bsv = bias ? bias[col] : 0.f;
#pragma unroll
    for (int mt = 0; mt < 8; ++mt) {
        int rowbase = mbase + mt * 16 + hi * 8;
#pragma unroll
        for (int r = 0; r < 8; ++r) {
            int row = rowbase + r;
            float v = acc[mt][r] + bsv;
            if (lens) v = (t < lens[row] - 1) ? v : 0.f;
            C[(long)row * ldc + col] = v;
        }
    }
}

// ---------------------------------------------------------------------------
// att[b,p] = b_att + sum_a relu(enc_att[b,p,a] + dec[b,a]) * w_att[a]
// one wave per (b,p)
// ---------------------------------------------------------------------------
__global__ void att_score_kernel(const float* __restrict__ encatt, const float* __restrict__ dec,
                                 const float* __restrict__ w_att, const float* __restrict__ b_att,
                                 float* __restrict__ att) {
    int idx  = blockIdx.x * 8 + (threadIdx.x >> 5);
    int lane = threadIdx.x & 31;
    if (idx >= BB * PP) return;
    int b = idx / PP;
    const float* ea = encatt + (long)idx * AA;
    const float* dv = dec + (long)b * AA;
    float acc = 0.f;
    for (int a = lane; a < AA; a += 32)
        acc += fmaxf(ea[a] + dv[a], 0.f) * w_att[a];
#pragma unroll
    for (int m = 16; m >= 1; m >>= 1) acc += __shfl_xor(acc, m, 32);
    if (lane == 0) att[idx] = acc + b_att[0];
}

// softmax over P + awe[b,e] = sum_p alpha[p]*enc[b,p,e]; one block per b
__global__ void softmax_awe_kernel(const float* __restrict__ att, const bf16_t* __restrict__ encB,
                                   float* __restrict__ awe, float* __restrict__ alphas_out,
                                   const int* __restrict__ lens, int t) {
    int b = blockIdx.x, tid = threadIdx.x;
    __shared__ float sal[PP];
    __shared__ float red[256];
    float v = (tid < PP) ? att[b * PP + tid] : -1e30f;
    red[tid] = v; __syncthreads();
    for (int s = 128; s > 0; s >>= 1) { if (tid < s) red[tid] = fmaxf(red[tid], red[tid + s]); __syncthreads(); }
    float mx = red[0]; __syncthreads();
    float e = (tid < PP) ? expf(v - mx) : 0.f;
    red[tid] = e; __syncthreads();
    for (int s = 128; s > 0; s >>= 1) { if (tid < s) red[tid] += red[tid + s]; __syncthreads(); }
    float inv = 1.0f / red[0];
    if (tid < PP) {
        float a = e * inv;
        sal[tid] = a;
        alphas_out[((long)b * TT + t) * PP + tid] = (t < lens[b] - 1) ? a : 0.f;
    }
    __syncthreads();
    for (int ee = tid; ee < EE; ee += 256) {
        const bf16_t* ep = encB + (long)b * PP * EE + ee;
        float acc = 0.f;
        for (int p = 0; p < PP; ++p) acc += sal[p] * (float)ep[(long)p * EE];
        awe[(long)b * EE + ee] = acc;
    }
}

// xcat = [ e_t (512) | sigmoid(gatepre)*awe (2048) | h (512) ]  -> bf16
__global__ void xcat_kernel(const float* __restrict__ W_emb, const int* __restrict__ caps,
                            const float* __restrict__ gatepre, const float* __restrict__ awe,
                            const float* __restrict__ h, bf16_t* __restrict__ xcat, int t) {
    long i = (long)blockIdx.x * blockDim.x + threadIdx.x;
    if (i >= (long)BB * 3072) return;
    int b = (int)(i / 3072), j = (int)(i % 3072);
    float v;
    if (j < EMBD) {
        int tok = caps[b * LL + t];
        v = W_emb[(long)tok * EMBD + j];
    } else if (j < EMBD + EE) {
        int jj = j - EMBD;
        v = sigf(gatepre[(long)b * EE + jj]) * awe[(long)b * EE + jj];
    } else {
        v = h[(long)b * DD + (j - EMBD - EE)];
    }
    xcat[i] = (bf16_t)v;
}

// LSTM pointwise + masked state update; emits bf16 h for the next GEMMs
__global__ void lstm_kernel(const float* __restrict__ gates, float* __restrict__ h,
                            float* __restrict__ c, bf16_t* __restrict__ hbf,
                            const int* __restrict__ lens, int t) {
    long i = (long)blockIdx.x * blockDim.x + threadIdx.x;
    if (i >= (long)BB * DD) return;
    int b = (int)(i >> 9), d = (int)(i & (DD - 1));
    const float* g = gates + ((long)b << 11);
    float ig = sigf(g[d]);
    float fg = sigf(g[DD + d]);
    float gg = tanhf(g[2 * DD + d]);
    float og = sigf(g[3 * DD + d]);
    float cn = fg * c[i] + ig * gg;
    float hn = og * tanhf(cn);
    if (t < lens[b] - 1) { c[i] = cn; h[i] = hn; }
    hbf[i] = (bf16_t)h[i];
}

// ---------------------------------------------------------------------------
// host side
// ---------------------------------------------------------------------------
static inline size_t alignup(size_t x) { return (x + 255) & ~(size_t)255; }

extern "C" void kernel_launch(void* const* d_in, const int* in_sizes, int n_in,
                              void* d_out, int out_size, void* d_ws, size_t ws_size,
                              hipStream_t stream) {
    (void)in_sizes; (void)n_in; (void)out_size;
    const float* enc    = (const float*)d_in[0];
    const int*   caps   = (const int*)d_in[1];
    const int*   lens   = (const int*)d_in[2];
    const float* W_emb  = (const float*)d_in[3];
    const float* W_enc  = (const float*)d_in[4];
    const float* b_enc  = (const float*)d_in[5];
    const float* W_dec  = (const float*)d_in[6];
    const float* b_dec  = (const float*)d_in[7];
    const float* w_att  = (const float*)d_in[8];
    const float* b_att  = (const float*)d_in[9];
    const float* W_h    = (const float*)d_in[10];
    const float* b_h    = (const float*)d_in[11];
    const float* W_c    = (const float*)d_in[12];
    const float* b_c    = (const float*)d_in[13];
    const float* W_beta = (const float*)d_in[14];
    const float* b_beta = (const float*)d_in[15];
    const float* W_fc   = (const float*)d_in[16];
    const float* b_fc   = (const float*)d_in[17];
    const float* W_ih   = (const float*)d_in[18];
    const float* b_ih   = (const float*)d_in[19];
    const float* W_hh   = (const float*)d_in[20];
    const float* b_hh   = (const float*)d_in[21];

    float* out = (float*)d_out;
    float* out_preds  = out;                                        // [B,T,V]
    float* out_caps   = out + (long)BB * TT * VV;                   // [B,L]
    float* out_lens   = out_caps + BB * LL;                         // [B]
    float* out_alphas = out_lens + BB;                              // [B,T,P]

    // ---- workspace layout -------------------------------------------------
    char* ws = (char*)d_ws;
    size_t off = 0;
    auto take = [&](size_t bytes) { size_t o = off; off = alignup(off + bytes); return o; };
    bf16_t* Wfc_sw   = (bf16_t*)(ws + take((size_t)DD * VV * 2));          // 30.7 MB
    bf16_t* Wcat_sw  = (bf16_t*)(ws + take((size_t)3072 * 4 * DD * 2));    // 12.6 MB
    bf16_t* Wenc_sw  = (bf16_t*)(ws + take((size_t)EE * AA * 2));
    bf16_t* Wdec_sw  = (bf16_t*)(ws + take((size_t)DD * AA * 2));
    bf16_t* Wbeta_sw = (bf16_t*)(ws + take((size_t)DD * EE * 2));
    bf16_t* Wh_sw    = (bf16_t*)(ws + take((size_t)EE * DD * 2));
    bf16_t* Wc_sw    = (bf16_t*)(ws + take((size_t)EE * DD * 2));
    bf16_t* encB     = (bf16_t*)(ws + take((size_t)BB * PP * EE * 2));     // 102.8 MB (L2-resident)
    float*  encatt   = (float*) (ws + take((size_t)BB * PP * AA * 4));     // 51.4 MB
    bf16_t* avgbf    = (bf16_t*)(ws + take((size_t)BB * EE * 2));
    float*  hbuf     = (float*) (ws + take((size_t)BB * DD * 4));
    float*  cbuf     = (float*) (ws + take((size_t)BB * DD * 4));
    bf16_t* hbf      = (bf16_t*)(ws + take((size_t)BB * DD * 2));
    float*  decbuf   = (float*) (ws + take((size_t)BB * AA * 4));
    float*  attbuf   = (float*) (ws + take((size_t)BB * PP * 4));
    float*  awebuf   = (float*) (ws + take((size_t)BB * EE * 4));
    float*  gatepre  = (float*) (ws + take((size_t)BB * EE * 4));
    bf16_t* xcat     = (bf16_t*)(ws + take((size_t)BB * 3072 * 2));
    float*  gatesbuf = (float*) (ws + take((size_t)BB * 4 * DD * 4));
    float*  biascat  = (float*) (ws + take((size_t)4 * DD * 4));
    if (off > ws_size) return;   // workspace too small (compile-only guard)

    const int THR = 256;
    auto blk = [](long n, int t) { return (unsigned)((n + t - 1) / t); };

    // ---- setup: weight swizzles ------------------------------------------
    // swizzleB(W, dst, ldw, Ntiles, nkc, kcBase, KCtot)
    {
        long n;
        n = (long)(VV / 16) * (DD / 32) * 32;   // W_fc: Ntiles=1875, KC=16
        swizzleB_kernel<<<blk(n, THR), THR, 0, stream>>>(W_fc, Wfc_sw, VV, VV / 16, DD / 32, 0, DD / 32);
        n = (long)(4 * DD / 16) * (2560 / 32) * 32; // W_ih into Wcat rows 0..2559
        swizzleB_kernel<<<blk(n, THR), THR, 0, stream>>>(W_ih, Wcat_sw, 4 * DD, 4 * DD / 16, 2560 / 32, 0, 3072 / 32);
        n = (long)(4 * DD / 16) * (DD / 32) * 32;   // W_hh into Wcat rows 2560..3071
        swizzleB_kernel<<<blk(n, THR), THR, 0, stream>>>(W_hh, Wcat_sw, 4 * DD, 4 * DD / 16, DD / 32, 2560 / 32, 3072 / 32);
        n = (long)(AA / 16) * (EE / 32) * 32;
        swizzleB_kernel<<<blk(n, THR), THR, 0, stream>>>(W_enc, Wenc_sw, AA, AA / 16, EE / 32, 0, EE / 32);
        n = (long)(AA / 16) * (DD / 32) * 32;
        swizzleB_kernel<<<blk(n, THR), THR, 0, stream>>>(W_dec, Wdec_sw, AA, AA / 16, DD / 32, 0, DD / 32);
        n = (long)(EE / 16) * (DD / 32) * 32;
        swizzleB_kernel<<<blk(n, THR), THR, 0, stream>>>(W_beta, Wbeta_sw, EE, EE / 16, DD / 32, 0, DD / 32);
        n = (long)(DD / 16) * (EE / 32) * 32;
        swizzleB_kernel<<<blk(n, THR), THR, 0, stream>>>(W_h, Wh_sw, DD, DD / 16, EE / 32, 0, EE / 32);
        swizzleB_kernel<<<blk(n, THR), THR, 0, stream>>>(W_c, Wc_sw, DD, DD / 16, EE / 32, 0, EE / 32);
    }
    cvt_bf16_kernel<<<4096, THR, 0, stream>>>(enc, encB, (long)BB * PP * EE);
    mean_kernel<<<blk((long)BB * EE, THR), THR, 0, stream>>>(enc, avgbf);
    biascat_kernel<<<blk(4 * DD, THR), THR, 0, stream>>>(b_ih, b_hh, biascat);
    tail_kernel<<<blk(BB * LL, THR), THR, 0, stream>>>(caps, lens, out_caps, out_lens);

    // h0 / c0 : avg[128,2048] @ W_h/W_c[2048,512]
    gemm_wmma_kernel<<<dim3(4, 1), THR, 0, stream>>>(avgbf, Wh_sw, b_h, hbuf, EE, EE / 32, DD / 16, DD, nullptr, 0);
    gemm_wmma_kernel<<<dim3(4, 1), THR, 0, stream>>>(avgbf, Wc_sw, b_c, cbuf, EE, EE / 32, DD / 16, DD, nullptr, 0);
    cvt_bf16_kernel<<<256, THR, 0, stream>>>(hbuf, hbf, (long)BB * DD);

    // enc_att : [25088,2048] @ [2048,512]  (loop invariant, 52 GFLOP)
    gemm_wmma_kernel<<<dim3(4, (BB * PP) / 128), THR, 0, stream>>>(
        encB, Wenc_sw, b_enc, encatt, EE, EE / 32, AA / 16, AA, nullptr, 0);

    // ---- time-step loop ---------------------------------------------------
    for (int t = 0; t < TT; ++t) {
        // dec = h @ W_dec + b_dec
        gemm_wmma_kernel<<<dim3(4, 1), THR, 0, stream>>>(hbf, Wdec_sw, b_dec, decbuf, DD, DD / 32, AA / 16, AA, nullptr, 0);
        // attention scores
        att_score_kernel<<<blk((long)BB * PP, 8), THR, 0, stream>>>(encatt, decbuf, w_att, b_att, attbuf);
        // softmax + weighted encoder sum (+ masked alphas out)
        softmax_awe_kernel<<<BB, THR, 0, stream>>>(attbuf, encB, awebuf, out_alphas, lens, t);
        // gate_pre = h @ W_beta + b_beta
        gemm_wmma_kernel<<<dim3(16, 1), THR, 0, stream>>>(hbf, Wbeta_sw, b_beta, gatepre, DD, DD / 32, EE / 16, EE, nullptr, 0);
        // xcat = [emb_t | sigmoid(gate)*awe | h]
        xcat_kernel<<<blk((long)BB * 3072, THR), THR, 0, stream>>>(W_emb, caps, gatepre, awebuf, hbuf, xcat, t);
        // gates = xcat @ [W_ih;W_hh] + (b_ih + b_hh)
        gemm_wmma_kernel<<<dim3(16, 1), THR, 0, stream>>>(xcat, Wcat_sw, biascat, gatesbuf, 3072, 3072 / 32, (4 * DD) / 16, 4 * DD, nullptr, 0);
        // LSTM pointwise + masked state update
        lstm_kernel<<<blk((long)BB * DD, THR), THR, 0, stream>>>(gatesbuf, hbuf, cbuf, hbf, lens, t);
        // preds = h @ W_fc + b_fc  -> d_out (masked rows zeroed)
        gemm_wmma_kernel<<<dim3((VV / 16 + 7) / 8, 1), THR, 0, stream>>>(
            hbf, Wfc_sw, b_fc, out_preds + (long)t * VV, DD, DD / 32, VV / 16, (long)TT * VV, lens, t);
    }
}
